// RandomizedMessagePassingLayer_43181601194854
// MI455X (gfx1250) — matrix operand order repaired and checked by reading, not compiled, not run on hardware
//
#include <hip/hip_runtime.h>

// ---------------------------------------------------------------------------
// GCN-style message passing on MI455X (gfx1250, wave32, WMMA).
//   h' = swish( (adj @ h) @ W + b ),   applied twice.
// adj: [B,N,N] f32 (binary), h: [B,N,U] (kept bf16 between steps),
// W: [U,U] f32, b: [U] f32.  B=8, N=4096, U=128.
// ---------------------------------------------------------------------------

#define BATCH   8
#define NNODES  4096
#define UDIM    128
#define TM      128      // rows per block
#define KT      64       // K chunk staged in LDS per iteration

typedef __attribute__((ext_vector_type(16))) __bf16 v16bf;
typedef __attribute__((ext_vector_type(8)))  float  v8f;

// --- helpers ----------------------------------------------------------------

// Assemble a v16bf operand from two 16-byte chunks (each 8 packed bf16).
__device__ inline v16bf load2x16(const void* p0, const void* p1) {
    union { uint4 q[2]; v16bf v; } u;
    u.q[0] = *(const uint4*)p0;
    u.q[1] = *(const uint4*)p1;
    return u.v;
}

// A operand for v_wmma_f32_16x16x32_bf16, built from a row of fp32 adj.
// Per ISA layout: lane holds K = koff..koff+7 (elems 0..7) and
// K = koff+16..koff+23 (elems 8..15); caller folds the lane-half offset
// into koff.
__device__ inline v16bf load_a_adj(const float* __restrict__ rowp, int koff) {
    float4 f0 = *(const float4*)(rowp + koff);
    float4 f1 = *(const float4*)(rowp + koff + 4);
    float4 f2 = *(const float4*)(rowp + koff + 16);
    float4 f3 = *(const float4*)(rowp + koff + 20);
    v16bf a;
    a[0]  = (__bf16)f0.x; a[1]  = (__bf16)f0.y; a[2]  = (__bf16)f0.z; a[3]  = (__bf16)f0.w;
    a[4]  = (__bf16)f1.x; a[5]  = (__bf16)f1.y; a[6]  = (__bf16)f1.z; a[7]  = (__bf16)f1.w;
    a[8]  = (__bf16)f2.x; a[9]  = (__bf16)f2.y; a[10] = (__bf16)f2.z; a[11] = (__bf16)f2.w;
    a[12] = (__bf16)f3.x; a[13] = (__bf16)f3.y; a[14] = (__bf16)f3.z; a[15] = (__bf16)f3.w;
    return a;
}

// --- elementwise fp32 -> bf16 conversion ------------------------------------

__global__ void to_bf16_kernel(const float* __restrict__ x,
                               __bf16* __restrict__ y, int n) {
    int i = blockIdx.x * blockDim.x + threadIdx.x;
    if (i < n) y[i] = (__bf16)x[i];
}

// --- one message-passing step ------------------------------------------------
// Block: 256 threads (8 waves). Grid: (N/TM, B).
// Wave w computes rows [rowBase + 16w, +16) x all 128 cols.

__global__ __launch_bounds__(256)
void gcn_step_kernel(const float*  __restrict__ adj,   // [B,N,N]
                     const __bf16* __restrict__ h_in,  // [B,N,U] bf16
                     const float*  __restrict__ Wm,    // [U,U]
                     const float*  __restrict__ bias,  // [U]
                     __bf16*       __restrict__ out_bf,  // nullable
                     float*        __restrict__ out_f32) // nullable
{
    // Region 1 (8704 dwords): main loop -> h chunk as column-major (k,k+1)
    // bf16 pairs [128 cols][36 dword stride]; epilogue -> per-wave hn tiles
    // bf16 row-major [16 rows][136 stride].
    // Region 2 (8704 dwords): W bf16 column-major [128 cols][136 stride].
    __shared__ unsigned int smem[17408];
    unsigned int* ldsH  = smem;
    __bf16*       ldsHN = (__bf16*)smem;
    __bf16*       ldsW  = (__bf16*)(smem + 8704);

    const int t    = threadIdx.x;
    const int lane = t & 31;
    const int wave = t >> 5;
    const int n16  = lane & 15;   // N index within 16x16 tile
    const int hlf  = lane >> 4;   // lane half (A/B K-half select)

    const int batch   = blockIdx.y;
    const int rowBase = blockIdx.x * TM;
    const int waveRow = rowBase + wave * 16;

    const float*  adjB = adj  + (size_t)batch * NNODES * NNODES;
    const __bf16* hB   = h_in + (size_t)batch * NNODES * UDIM;
    const float*  aRow = adjB + (size_t)(waveRow + n16) * NNODES;

    v8f acc[8];
#pragma unroll
    for (int i = 0; i < 8; ++i) acc[i] = {};

    // ---- main loop: hn = adj @ h -------------------------------------------
    for (int k0 = 0; k0 < NNODES; k0 += KT) {
        __syncthreads();
        // Stage h[k0:k0+64, 0:128] into LDS, column-major packed pairs.
        // 512 units: unit = (pair-row 0..31) x (col-octet 0..15).
#pragma unroll
        for (int u = 0; u < 2; ++u) {
            int unit = t + u * 256;
            int prow = unit & 31;
            int c8   = (unit >> 5) * 8;
            const __bf16* g0 = hB + (size_t)(k0 + 2 * prow) * UDIM + c8;
            uint4 ra = *(const uint4*)g0;            // row k,   8 bf16
            uint4 rb = *(const uint4*)(g0 + UDIM);   // row k+1, 8 bf16
            unsigned int ua[4] = {ra.x, ra.y, ra.z, ra.w};
            unsigned int ub[4] = {rb.x, rb.y, rb.z, rb.w};
#pragma unroll
            for (int i = 0; i < 4; ++i) {
                unsigned int lo = (ua[i] & 0xFFFFu) | (ub[i] << 16);         // col c+2i
                unsigned int hi = (ua[i] >> 16)     | (ub[i] & 0xFFFF0000u); // col c+2i+1
                ldsH[(c8 + 2 * i)     * 36 + prow] = lo;
                ldsH[(c8 + 2 * i + 1) * 36 + prow] = hi;
            }
        }
        __syncthreads();

#pragma unroll
        for (int kk = 0; kk < KT; kk += 32) {
            v16bf a = load_a_adj(aRow, k0 + kk + hlf * 8);
#pragma unroll
            for (int ct = 0; ct < 8; ++ct) {
                const unsigned int* bp =
                    ldsH + (ct * 16 + n16) * 36 + (kk >> 1) + hlf * 4;
                v16bf b = load2x16(bp, bp + 8);
                acc[ct] = __builtin_amdgcn_wmma_f32_16x16x32_bf16(
                    false, a, false, b, (short)0, acc[ct], false, false);
            }
        }
    }
    __syncthreads();   // all waves done reading ldsH before it is reused

    // ---- epilogue: out = swish(hn @ W + b) ---------------------------------
    // Spill hn (C layout) to LDS as bf16 row-major, per-wave region.
    __bf16* hnW = ldsHN + wave * (16 * 136);
#pragma unroll
    for (int ct = 0; ct < 8; ++ct)
#pragma unroll
        for (int r = 0; r < 8; ++r)
            hnW[(r + 8 * hlf) * 136 + ct * 16 + n16] = (__bf16)acc[ct][r];

    // Stage W (fp32 row-major) -> bf16 column-major in LDS.
    for (int idx = t; idx < 128 * 32; idx += 256) {
        int k  = idx >> 5;
        int c4 = (idx & 31) * 4;
        float4 w4 = *(const float4*)(Wm + k * 128 + c4);
        ldsW[(c4 + 0) * 136 + k] = (__bf16)w4.x;
        ldsW[(c4 + 1) * 136 + k] = (__bf16)w4.y;
        ldsW[(c4 + 2) * 136 + k] = (__bf16)w4.z;
        ldsW[(c4 + 3) * 136 + k] = (__bf16)w4.w;
    }
    __syncthreads();

    v8f acc2[8];
#pragma unroll
    for (int i = 0; i < 8; ++i) acc2[i] = {};

#pragma unroll
    for (int k0 = 0; k0 < UDIM; k0 += 32) {
        const __bf16* ap = hnW + n16 * 136 + k0 + hlf * 8;
        v16bf a = load2x16(ap, ap + 16);          // K 0..7 and K 16..23 (+half)
#pragma unroll
        for (int ct = 0; ct < 8; ++ct) {
            const __bf16* bp = ldsW + (ct * 16 + n16) * 136 + k0 + hlf * 8;
            v16bf b = load2x16(bp, bp + 16);      // K 0..7 and K 16..23 (+half)
            acc2[ct] = __builtin_amdgcn_wmma_f32_16x16x32_bf16(
                false, a, false, b, (short)0, acc2[ct], false, false);
        }
    }

    // Bias + swish + store.
#pragma unroll
    for (int ct = 0; ct < 8; ++ct) {
        float bv = bias[ct * 16 + n16];
#pragma unroll
        for (int r = 0; r < 8; ++r) {
            float v = acc2[ct][r] + bv;
            float s = v / (1.0f + __expf(-v));    // swish
            int gRow = waveRow + r + 8 * hlf;
            size_t oidx = ((size_t)batch * NNODES + gRow) * UDIM + ct * 16 + n16;
            if (out_bf)  out_bf[oidx]  = (__bf16)s;
            if (out_f32) out_f32[oidx] = s;
        }
    }
}

// ---------------------------------------------------------------------------

extern "C" void kernel_launch(void* const* d_in, const int* in_sizes, int n_in,
                              void* d_out, int out_size, void* d_ws, size_t ws_size,
                              hipStream_t stream) {
    const float* x   = (const float*)d_in[0];   // [B,N,U]
    const float* adj = (const float*)d_in[1];   // [B,N,N]
    const float* Wm  = (const float*)d_in[2];   // [U,U]
    const float* b   = (const float*)d_in[3];   // [U]
    float* out = (float*)d_out;

    const int nElem = BATCH * NNODES * UDIM;    // 4,194,304
    __bf16* h0 = (__bf16*)d_ws;                                   // 8 MB
    __bf16* h1 = (__bf16*)((char*)d_ws + (size_t)nElem * 2);      // 8 MB

    to_bf16_kernel<<<(nElem + 255) / 256, 256, 0, stream>>>(x, h0, nElem);

    dim3 grid(NNODES / TM, BATCH);   // (32, 8)
    gcn_step_kernel<<<grid, 256, 0, stream>>>(adj, h0, Wm, b, h1, nullptr);
    gcn_step_kernel<<<grid, 256, 0, stream>>>(adj, h1, Wm, b, nullptr, out);
}
